// PointRendRefiner_72688026517970
// MI455X (gfx1250) — matrix operand order repaired and checked by reading, not compiled, not run on hardware
//
#include <hip/hip_runtime.h>

typedef __attribute__((ext_vector_type(16))) _Float16 v16h;
typedef __attribute__((ext_vector_type(8)))  float    v8f;
typedef unsigned int u32x4 __attribute__((ext_vector_type(4)));
typedef int          i32x4 __attribute__((ext_vector_type(4)));
typedef int          i32x8 __attribute__((ext_vector_type(8)));

#define BATCH 4
#define HC 256
#define NBINS 65536
#define TS 32          // dilate tile (output) side
#define CHUNK 64       // eq-compaction chunk

#if defined(__HIP_DEVICE_COMPILE__) && __has_builtin(__builtin_amdgcn_tensor_load_to_lds)
#define HAVE_TDM 1
#else
#define HAVE_TDM 0
#endif

static inline int ceil_div(int a, int b) { return (a + b - 1) / b; }

// out layout helper: mode 0 = contiguous [B,4,L,L]; mode 1 = final split
// [B,3,L,L] (fgr) followed by [B,1,L,L] (pha), i.e. d_out layout.
__device__ __forceinline__ size_t out_off(int b, int c, int p, int L, int mode) {
    size_t HW = (size_t)L * (size_t)L;
    if (mode == 0) return ((size_t)(b * 4 + c)) * HW + (size_t)p;
    return (c < 3) ? ((size_t)(b * 3 + c) * HW + (size_t)p)
                   : ((size_t)(BATCH * 3) * HW + (size_t)b * HW + (size_t)p);
}

// ---------------------------------------------------------------- concat
__global__ void k_concat(const float* __restrict__ fgr,
                         const float* __restrict__ pha,
                         float* __restrict__ out) {
    int gid = blockIdx.x * blockDim.x + threadIdx.x;
    const int HW = HC * HC;
    int total = BATCH * 4 * HW;
    if (gid >= total) return;
    int p = gid % HW;
    int c = (gid / HW) % 4;
    int b = gid / (HW * 4);
    out[gid] = (c < 3) ? fgr[((size_t)(b * 3 + c)) * HW + p]
                       : pha[(size_t)b * HW + p];
}

// ------------------------------------------- bilinear x2, align_corners=T
__global__ void k_upsample2(const float* __restrict__ in, float* __restrict__ out,
                            int C, int H, int W, int mode) {
    int OH = 2 * H, OW = 2 * W;
    long long gid = (long long)blockIdx.x * blockDim.x + threadIdx.x;
    long long total = (long long)BATCH * C * OH * OW;
    if (gid >= total) return;
    int ox = (int)(gid % OW);
    long long t = gid / OW;
    int oy = (int)(t % OH);
    t /= OH;
    int c = (int)(t % C);
    int b = (int)(t / C);

    float px = (float)ox * (float)(W - 1) / (float)(OW - 1);
    float py = (float)oy * (float)(H - 1) / (float)(OH - 1);
    int ix0 = (int)floorf(px), iy0 = (int)floorf(py);
    int ix1 = min(ix0 + 1, W - 1), iy1 = min(iy0 + 1, H - 1);
    float fx = px - (float)ix0, fy = py - (float)iy0;

    const float* base = in + ((size_t)(b * C + c)) * ((size_t)H * W);
    float v00 = base[(size_t)iy0 * W + ix0];
    float v01 = base[(size_t)iy0 * W + ix1];
    float v10 = base[(size_t)iy1 * W + ix0];
    float v11 = base[(size_t)iy1 * W + ix1];
    float v = (v00 * (1.f - fx) + v01 * fx) * (1.f - fy) +
              (v10 * (1.f - fx) + v11 * fx) * fy;

    if (mode == 0) out[gid] = v;
    else           out[out_off(b, c, oy * OW + ox, OW, 1)] = v;
}

// ---------------- 5x5 ellipse dilation, LDS tile staged via TDM async DMA
__global__ __launch_bounds__(256) void k_dilate(const float* __restrict__ e,
                                                float* __restrict__ unc, int L) {
    __shared__ float smem[(TS + 4) * (TS + 4)];
    int tilesx = L / TS;
    int bid = blockIdx.x;
    int b    = bid / (tilesx * tilesx);
    int trow = (bid / tilesx) % tilesx;
    int tcol = bid % tilesx;
    int ty0 = trow * TS, tx0 = tcol * TS;
    int cy0 = max(ty0 - 2, 0), cx0 = max(tx0 - 2, 0);
    int cy1 = min(ty0 + TS + 2, L), cx1 = min(tx0 + TS + 2, L);
    int cw = cx1 - cx0, ch = cy1 - cy0;     // clipped halo tile, fully in-bounds
    const float* src = e + (size_t)b * L * L;

#if HAVE_TDM
    if (threadIdx.x < 32) {   // wave 0 issues the DMA; TDM ignores EXEC
        unsigned long long ga =
            (unsigned long long)(uintptr_t)(src + (size_t)cy0 * L + cx0);
        unsigned lds = (unsigned)(unsigned long long)(uintptr_t)&smem[0];
        u32x4 g0;
        g0.x = 1u;                                       // count=1, load
        g0.y = lds;                                      // lds_addr (bytes)
        g0.z = (unsigned)(ga & 0xFFFFFFFFu);             // global_addr[31:0]
        g0.w = (unsigned)((ga >> 32) & 0x1FFFFFFu) | 0x80000000u; // [56:32] | type=2
        i32x8 g1;
        unsigned uL = (unsigned)L, ucw = (unsigned)cw, uch = (unsigned)ch;
        g1[0] = (int)(2u << 16);                         // data_size = 4 bytes
        g1[1] = (int)((uL & 0xFFFFu) << 16);             // tensor_dim0[15:0]
        g1[2] = (int)(((uL >> 16) & 0xFFFFu) | ((uL & 0xFFFFu) << 16));   // d0 hi | d1 lo
        g1[3] = (int)(((uL >> 16) & 0xFFFFu) | ((ucw & 0xFFFFu) << 16));  // d1 hi | tile_dim0
        g1[4] = (int)(uch & 0xFFFFu);                    // tile_dim1 (tile_dim2 = 0)
        g1[5] = (int)uL;                                 // tensor_dim0_stride[31:0]
        g1[6] = 0;
        g1[7] = 0;
        i32x4 gz = {0, 0, 0, 0};
#if defined(__clang_major__) && __clang_major__ >= 23
        i32x8 gz8 = {0, 0, 0, 0, 0, 0, 0, 0};
        __builtin_amdgcn_tensor_load_to_lds(g0, g1, gz, gz, gz8, 0);
#else
        __builtin_amdgcn_tensor_load_to_lds(g0, g1, gz, gz, 0);
#endif
        __builtin_amdgcn_s_wait_tensorcnt(0);
    }
    __syncthreads();
#else
    for (int i = threadIdx.x; i < cw * ch; i += 256)
        smem[i] = src[(size_t)(cy0 + i / cw) * L + (cx0 + i % cw)];
    __syncthreads();
#endif

    for (int i = threadIdx.x; i < TS * TS; i += 256) {
        int ly = i / TS, lx = i % TS;
        int y = ty0 + ly, x = tx0 + lx;
        float s = 0.f;
#pragma unroll
        for (int dy = -2; dy <= 2; ++dy) {
#pragma unroll
            for (int dx = -2; dx <= 2; ++dx) {
                bool on = (dy == -2 || dy == 2) ? (dx == 0) : true;
                if (!on) continue;
                int gy = y + dy, gx = x + dx;
                if (gy >= cy0 && gy < cy1 && gx >= cx0 && gx < cx1)
                    s += smem[(gy - cy0) * cw + (gx - cx0)];
            }
        }
        unc[((size_t)b * L + y) * L + x] = s;
    }
}

// ---------------------------------------------------------------- top-k
__global__ void k_zero_i32(int* p, int n) {
    int g = blockIdx.x * blockDim.x + threadIdx.x;
    if (g < n) p[g] = 0;
}

__global__ void k_histA(const float* __restrict__ unc, int* __restrict__ hist, int HW) {
    long long gid = (long long)blockIdx.x * blockDim.x + threadIdx.x;
    if (gid >= (long long)BATCH * HW) return;
    int b = (int)(gid / HW);
    unsigned bits = __float_as_uint(unc[gid]);   // values >= 0 -> monotonic
    atomicAdd(&hist[b * NBINS + (int)(bits >> 16)], 1);
}

// block-parallel descending scan over 64K bins: 256 partials + short serial tail
__global__ __launch_bounds__(256) void k_scanA(const int* __restrict__ hist,
                                               int* prefix1, int* r1, int k) {
    __shared__ int part[256];
    int b = blockIdx.x, t = threadIdx.x;
    const int* h = hist + b * NBINS;
    int hi = NBINS - 1 - 256 * t;
    int s = 0;
    for (int i = 0; i < 256; ++i) s += h[hi - i];
    part[t] = s;
    __syncthreads();
    if (t == 0) {
        int cum = 0;
        for (int seg = 0; seg < 256; ++seg) {
            if (cum + part[seg] >= k) {
                int bh = NBINS - 1 - 256 * seg;
                for (int i = 0; i < 256; ++i) {
                    int c = h[bh - i];
                    if (cum + c >= k) { prefix1[b] = bh - i; r1[b] = k - cum; return; }
                    cum += c;
                }
            }
            cum += part[seg];
        }
        prefix1[b] = 0; r1[b] = k - cum;
    }
}

__global__ void k_histB(const float* __restrict__ unc, int* __restrict__ hist,
                        const int* __restrict__ prefix1, int HW) {
    long long gid = (long long)blockIdx.x * blockDim.x + threadIdx.x;
    if (gid >= (long long)BATCH * HW) return;
    int b = (int)(gid / HW);
    unsigned bits = __float_as_uint(unc[gid]);
    if ((int)(bits >> 16) == prefix1[b])
        atomicAdd(&hist[b * NBINS + (int)(bits & 0xFFFFu)], 1);
}

__global__ __launch_bounds__(256) void k_scanB(const int* __restrict__ hist,
                                               const int* __restrict__ prefix1,
                                               const int* __restrict__ r1,
                                               unsigned* vbits, int* rsel) {
    __shared__ int part[256];
    int b = blockIdx.x, t = threadIdx.x;
    const int* h = hist + b * NBINS;
    int target = r1[b];
    int hi = NBINS - 1 - 256 * t;
    int s = 0;
    for (int i = 0; i < 256; ++i) s += h[hi - i];
    part[t] = s;
    __syncthreads();
    if (t == 0) {
        int cum = 0;
        for (int seg = 0; seg < 256; ++seg) {
            if (cum + part[seg] >= target) {
                int bh = NBINS - 1 - 256 * seg;
                for (int i = 0; i < 256; ++i) {
                    int c = h[bh - i];
                    if (cum + c >= target) {
                        vbits[b] = ((unsigned)prefix1[b] << 16) | (unsigned)(bh - i);
                        rsel[b] = target - cum;
                        return;
                    }
                    cum += c;
                }
            }
            cum += part[seg];
        }
        vbits[b] = (unsigned)prefix1[b] << 16; rsel[b] = target - cum;
    }
}

__global__ void k_compact_gt(const float* __restrict__ unc, const unsigned* __restrict__ vbits,
                             int* __restrict__ cnt, int* __restrict__ idxbuf, int HW, int k) {
    long long gid = (long long)blockIdx.x * blockDim.x + threadIdx.x;
    if (gid >= (long long)BATCH * HW) return;
    int b = (int)(gid / HW);
    int p = (int)(gid % HW);
    if (__float_as_uint(unc[gid]) > vbits[b]) {
        int pos = atomicAdd(&cnt[b], 1);
        if (pos < k) idxbuf[(size_t)b * k + pos] = p;
    }
}

// --- deterministic eq-compaction: tied values taken lowest-index-first ---
__global__ void k_count_eq(const float* __restrict__ unc, const unsigned* __restrict__ vbits,
                           int* __restrict__ ccnt, int HW) {
    int nch = HW / CHUNK;
    int g = blockIdx.x * blockDim.x + threadIdx.x;
    if (g >= BATCH * nch) return;
    int b = g / nch, j = g % nch;
    unsigned v = vbits[b];
    const float* u = unc + (size_t)b * HW + (size_t)j * CHUNK;
    int c = 0;
    for (int i = 0; i < CHUNK; ++i) c += (__float_as_uint(u[i]) == v) ? 1 : 0;
    ccnt[g] = c;
}

__global__ void k_scan_eq(const int* __restrict__ ccnt, int* __restrict__ coff, int HW) {
    int b = blockIdx.x * blockDim.x + threadIdx.x;
    if (b >= BATCH) return;
    int nch = HW / CHUNK, acc = 0;
    for (int j = 0; j < nch; ++j) { coff[b * nch + j] = acc; acc += ccnt[b * nch + j]; }
}

__global__ void k_scatter_eq(const float* __restrict__ unc, const unsigned* __restrict__ vbits,
                             const int* __restrict__ rsel, const int* __restrict__ coff,
                             int* __restrict__ idxbuf, int HW, int k) {
    int nch = HW / CHUNK;
    int g = blockIdx.x * blockDim.x + threadIdx.x;
    if (g >= BATCH * nch) return;
    int b = g / nch, j = g % nch;
    unsigned v = vbits[b];
    int r = rsel[b];
    int rank = coff[g];
    if (rank >= r) return;
    int base = k - r;
    const float* u = unc + (size_t)b * HW + (size_t)j * CHUNK;
    for (int i = 0; i < CHUNK && rank < r; ++i) {
        if (__float_as_uint(u[i]) == v) {
            idxbuf[(size_t)b * k + base + rank] = j * CHUNK + i;
            ++rank;
        }
    }
}

// ------------------------------------ WMMA point MLP (D = W[16x32]·F[32x16])
__device__ __forceinline__ float sample_hid(const float* __restrict__ hid, int b, int c,
                                            int x0, int y0, float wx, float wy) {
    const int HW = HC * HC;
    const float* base = hid + ((size_t)(b * 16 + c)) * HW;
    int x1 = x0 + 1, y1 = y0 + 1;
    float v00 = (x0 >= 0 && x0 < HC && y0 >= 0 && y0 < HC) ? base[(size_t)y0 * HC + x0] : 0.f;
    float v01 = (x1 >= 0 && x1 < HC && y0 >= 0 && y0 < HC) ? base[(size_t)y0 * HC + x1] : 0.f;
    float v10 = (x0 >= 0 && x0 < HC && y1 >= 0 && y1 < HC) ? base[(size_t)y1 * HC + x0] : 0.f;
    float v11 = (x1 >= 0 && x1 < HC && y1 >= 0 && y1 < HC) ? base[(size_t)y1 * HC + x1] : 0.f;
    float top = v00 * (1.f - wx) + v01 * wx;
    float bot = v10 * (1.f - wx) + v11 * wx;
    return top * (1.f - wy) + bot * wy;
}

__global__ __launch_bounds__(32) void k_refine(
    const float* __restrict__ hid, const float* __restrict__ mlp_w,
    const float* __restrict__ mlp_b, const int* __restrict__ idxbuf,
    float* __restrict__ outbuf, int L, int k, int mode) {
    int lane = threadIdx.x;             // wave32
    int tile = blockIdx.x;              // B * k/16 tiles
    int tiles_per_b = k / 16;
    int b = tile / tiles_per_b;
    int t = tile % tiles_per_b;
    const int* tidx = idxbuf + (size_t)b * k + (size_t)t * 16;

    // ---- A matrix: weights, 16x32 f16 (rows 4..15 and K 20..31 zero)
    v16h A;
    {
        int m = lane & 15;
        int kbase = (lane < 16) ? 0 : 8;
#pragma unroll
        for (int h = 0; h < 16; ++h) {
            int kk = (h < 8) ? (kbase + h) : (kbase + 8 + h);
            float wv = (m < 4 && kk < 20) ? mlp_w[m * 20 + kk] : 0.f;
            A[h] = (_Float16)wv;
        }
    }

    // ---- B matrix: features, 32x16 f16; col = lane&15
    int n = lane & 15;
    int p = tidx[n];
    int px = p % L, py = p / L;
    float xs = ((float)px + 0.5f) * ((float)HC / (float)L) - 0.5f;
    float ys = ((float)py + 0.5f) * ((float)HC / (float)L) - 0.5f;
    float x0f = floorf(xs), y0f = floorf(ys);
    float wx = xs - x0f, wy = ys - y0f;
    int x0 = (int)x0f, y0 = (int)y0f;

    v16h Bm;
#pragma unroll
    for (int h = 0; h < 16; ++h) {
        int kk = (h < 8) ? (((lane < 16) ? 0 : 8) + h) : (((lane < 16) ? 0 : 8) + 8 + h);
        float v;
        if (kk < 4)       v = outbuf[out_off(b, kk, p, L, mode)];      // exact pixel
        else if (kk < 20) v = sample_hid(hid, b, kk - 4, x0, y0, wx, wy);
        else              v = 0.f;
        Bm[h] = (_Float16)v;
    }

    v8f C = {};
    C = __builtin_amdgcn_wmma_f32_16x16x32_f16(false, A, false, Bm,
                                               (short)0, C, false, false);

    if (lane < 16) {
#pragma unroll
        for (int r = 0; r < 4; ++r) {
            float val = C[r] + mlp_b[r];
            outbuf[out_off(b, r, p, L, mode)] = val;
        }
    }
}

// ----------------------------------------------------------------- host
static void select_topk(hipStream_t s, const float* unc, int HW, int k,
                        int* hist, int* prefix1, int* r1, unsigned* vbits,
                        int* rsel, int* cnt, int* ccnt, int* coff, int* idxbuf) {
    int nh = BATCH * NBINS;
    long long npix = (long long)BATCH * HW;
    int pxblocks = (int)((npix + 255) / 256);
    int nch = HW / CHUNK;
    int chblocks = ceil_div(BATCH * nch, 256);
    k_zero_i32<<<ceil_div(nh, 256), 256, 0, s>>>(hist, nh);
    k_zero_i32<<<1, 32, 0, s>>>(cnt, BATCH);
    k_histA<<<pxblocks, 256, 0, s>>>(unc, hist, HW);
    k_scanA<<<BATCH, 256, 0, s>>>(hist, prefix1, r1, k);
    k_zero_i32<<<ceil_div(nh, 256), 256, 0, s>>>(hist, nh);
    k_histB<<<pxblocks, 256, 0, s>>>(unc, hist, prefix1, HW);
    k_scanB<<<BATCH, 256, 0, s>>>(hist, prefix1, r1, vbits, rsel);
    k_compact_gt<<<pxblocks, 256, 0, s>>>(unc, vbits, cnt, idxbuf, HW, k);
    k_count_eq<<<chblocks, 256, 0, s>>>(unc, vbits, ccnt, HW);
    k_scan_eq<<<1, 32, 0, s>>>(ccnt, coff, HW);
    k_scatter_eq<<<chblocks, 256, 0, s>>>(unc, vbits, rsel, coff, idxbuf, HW, k);
}

extern "C" void kernel_launch(void* const* d_in, const int* in_sizes, int n_in,
                              void* d_out, int out_size, void* d_ws, size_t ws_size,
                              hipStream_t stream) {
    (void)in_sizes; (void)n_in; (void)out_size; (void)ws_size;
    const float* hid   = (const float*)d_in[1];
    const float* fgr   = (const float*)d_in[2];
    const float* pha   = (const float*)d_in[3];
    const float* err   = (const float*)d_in[4];
    const float* mlp_w = (const float*)d_in[5];
    const float* mlp_b = (const float*)d_in[6];
    float* out_final = (float*)d_out;

    const int HW256 = 256 * 256, HW512 = 512 * 512, HW1024 = 1024 * 1024;

    char* w = (char*)d_ws;
    auto carve = [&](size_t bytes) -> void* {
        void* p = (void*)w;
        w += (bytes + 255) & ~(size_t)255;
        return p;
    };
    float*    out256  = (float*)carve((size_t)BATCH * 4 * HW256 * 4);
    float*    out512  = (float*)carve((size_t)BATCH * 4 * HW512 * 4);
    float*    err512  = (float*)carve((size_t)BATCH * HW512 * 4);
    float*    err1024 = (float*)carve((size_t)BATCH * HW1024 * 4);
    float*    unc     = (float*)carve((size_t)BATCH * HW1024 * 4);
    int*      hist    = (int*)carve((size_t)BATCH * NBINS * 4);
    int*      prefix1 = (int*)carve(BATCH * 4);
    int*      r1      = (int*)carve(BATCH * 4);
    unsigned* vbits   = (unsigned*)carve(BATCH * 4);
    int*      rsel    = (int*)carve(BATCH * 4);
    int*      cnt     = (int*)carve(BATCH * 4);
    int*      ccnt    = (int*)carve((size_t)BATCH * (HW1024 / CHUNK) * 4);
    int*      coff    = (int*)carve((size_t)BATCH * (HW1024 / CHUNK) * 4);
    int*      idxbuf  = (int*)carve((size_t)BATCH * 16384 * 4);

    hipStream_t s = stream;

    k_concat<<<ceil_div(BATCH * 4 * HW256, 256), 256, 0, s>>>(fgr, pha, out256);

    // ---------------- level 1: 256 -> 512, k = 64*64
    k_upsample2<<<ceil_div(BATCH * 4 * HW512, 256), 256, 0, s>>>(out256, out512, 4, 256, 256, 0);
    k_upsample2<<<ceil_div(BATCH * 1 * HW512, 256), 256, 0, s>>>(err, err512, 1, 256, 256, 0);
    k_dilate<<<BATCH * (512 / TS) * (512 / TS), 256, 0, s>>>(err512, unc, 512);
    select_topk(s, unc, HW512, 4096, hist, prefix1, r1, vbits, rsel, cnt, ccnt, coff, idxbuf);
    k_refine<<<BATCH * 4096 / 16, 32, 0, s>>>(hid, mlp_w, mlp_b, idxbuf, out512, 512, 4096, 0);

    // ---------------- level 2: 512 -> 1024 (direct into d_out split layout)
    k_upsample2<<<ceil_div(BATCH * 4 * HW1024, 256), 256, 0, s>>>(out512, out_final, 4, 512, 512, 1);
    k_upsample2<<<ceil_div(BATCH * 1 * HW1024, 256), 256, 0, s>>>(err512, err1024, 1, 512, 512, 0);
    k_dilate<<<BATCH * (1024 / TS) * (1024 / TS), 256, 0, s>>>(err1024, unc, 1024);
    select_topk(s, unc, HW1024, 16384, hist, prefix1, r1, vbits, rsel, cnt, ccnt, coff, idxbuf);
    k_refine<<<BATCH * 16384 / 16, 32, 0, s>>>(hid, mlp_w, mlp_b, idxbuf, out_final, 1024, 16384, 1);
}
